// KernelWindowAttention_85341000171912
// MI455X (gfx1250) — matrix operand order repaired and checked by reading, not compiled, not run on hardware
//
#include <hip/hip_runtime.h>

typedef __attribute__((ext_vector_type(16))) _Float16 v16h;
typedef __attribute__((ext_vector_type(8)))  _Float16 v8h;
typedef __attribute__((ext_vector_type(8)))  float    v8f;
typedef __attribute__((ext_vector_type(4)))  float    v4f;

#define WMMA_F16(a, b, c) \
  __builtin_amdgcn_wmma_f32_16x16x32_f16(false, (a), false, (b), (short)0, (c), false, false)

constexpr int Bb    = 512;
constexpr int Nn    = 64;
constexpr int DIM   = 512;
constexpr int Hh    = 8;
constexpr int HD    = 64;
constexpr int Mrows = Bb * Nn; // 32768

// ---------------------------------------------------------------------------
// Weight prep: fp32 -> f16, transposed (WT[n][k] = W[k][n]) so B-fragments are
// contiguous-K loads.
// ---------------------------------------------------------------------------
__global__ void prep_weights(const float* __restrict__ qw,
                             const float* __restrict__ kvw,
                             const float* __restrict__ pw,
                             _Float16* __restrict__ qwT,
                             _Float16* __restrict__ kwT,
                             _Float16* __restrict__ vwT,
                             _Float16* __restrict__ pwT) {
  int idx = blockIdx.x * blockDim.x + threadIdx.x; // idx = n*512 + k
  int n = idx >> 9;
  int k = idx & 511;
  qwT[idx] = (_Float16)qw[k * DIM + n];
  pwT[idx] = (_Float16)pw[k * DIM + n];
  kwT[idx] = (_Float16)kvw[k * (2 * DIM) + n];
  vwT[idx] = (_Float16)kvw[k * (2 * DIM) + DIM + n];
}

// ---------------------------------------------------------------------------
// GEMM: C = A[M][K] * WT^T + bias. One 64x64 tile per wave, 8 waves/block.
// All waves of a block share the same 64 A-rows (tilesN==8): the A panel is
// staged once per k-step into LDS (f16), cutting A traffic 8x.
// HEADT epilogue writes head-transposed f16: out[(b*8+h)*64+d][n] with
// contiguous-n v8h stores (used for qT / kT feeding attention fragments).
// ---------------------------------------------------------------------------
template <typename AT, typename OT, bool HEADT>
__global__ __launch_bounds__(256) void gemm64(const AT* __restrict__ A,
                                              const _Float16* __restrict__ WT,
                                              const float* __restrict__ bias,
                                              OT* __restrict__ C,
                                              int M, int N, int K) {
  __shared__ alignas(16) _Float16 sA[64][48]; // 64 rows x 32 k (+16 pad)
  const int tid  = threadIdx.x;
  const int lane = tid & 31;
  const int wid  = tid >> 5;
  const int l15  = lane & 15;
  const int half = lane >> 4;
  const int m0 = blockIdx.x << 6; // shared by all waves in the block
  const int n0 = wid << 6;        // tilesN == 8 == waves per block

  const int ldr = tid >> 2;        // staging: row 0..63
  const int ldg = (tid & 3) * 8;   // staging: k-group start {0,8,16,24}

  const v8f zero8 = {0.f, 0.f, 0.f, 0.f, 0.f, 0.f, 0.f, 0.f};
  v8f acc[4][4];
#pragma unroll
  for (int i = 0; i < 4; ++i)
#pragma unroll
    for (int j = 0; j < 4; ++j) acc[i][j] = zero8;

  for (int k0 = 0; k0 < K; k0 += 32) {
    __syncthreads(); // previous-iteration sA reads complete
    if constexpr (sizeof(AT) == 4) {
      const float* ap = (const float*)A + (size_t)(m0 + ldr) * K + k0 + ldg;
      v4f f0 = *(const v4f*)(ap);
      v4f f1 = *(const v4f*)(ap + 4);
      if (k0 + 32 < K) __builtin_prefetch(ap + 32, 0, 3);
      v8h hv;
#pragma unroll
      for (int j = 0; j < 4; ++j) {
        hv[j]     = (_Float16)f0[j];
        hv[4 + j] = (_Float16)f1[j];
      }
      *(v8h*)&sA[ldr][ldg] = hv;
    } else {
      const _Float16* ap = (const _Float16*)A + (size_t)(m0 + ldr) * K + k0 + ldg;
      unsigned lds_addr = (unsigned)(size_t)&sA[ldr][ldg];
      // CDNA5 async global->LDS copy (ASYNCcnt-tracked), 16B per lane.
      asm volatile("global_load_async_to_lds_b128 %0, %1, off"
                   :: "v"(lds_addr), "v"(ap) : "memory");
      if (k0 + 32 < K) __builtin_prefetch(ap + 32, 0, 3);
      asm volatile("s_wait_asynccnt 0" ::: "memory");
    }
    __syncthreads();

    // A fragments from LDS (vector b128 reads).
    v16h a[4], b[4];
#pragma unroll
    for (int mi = 0; mi < 4; ++mi) {
      const int r = mi * 16 + l15;
      v8h h0 = *(const v8h*)&sA[r][half * 16];
      v8h h1 = *(const v8h*)&sA[r][half * 16 + 8];
#pragma unroll
      for (int j = 0; j < 8; ++j) { a[mi][j] = h0[j]; a[mi][8 + j] = h1[j]; }
    }
    // B fragments from WT (row-major over n, contiguous k): lane half h holds
    // K = k0 + h*16 + 0..15, column = l15.
#pragma unroll
    for (int ni = 0; ni < 4; ++ni) {
      const _Float16* wp = WT + (size_t)(n0 + ni * 16 + l15) * K + k0 + half * 16;
      v8h w0 = *(const v8h*)(wp);
      v8h w1 = *(const v8h*)(wp + 8);
#pragma unroll
      for (int j = 0; j < 8; ++j) { b[ni][j] = w0[j]; b[ni][8 + j] = w1[j]; }
    }
#pragma unroll
    for (int mi = 0; mi < 4; ++mi)
#pragma unroll
      for (int ni = 0; ni < 4; ++ni)
        acc[mi][ni] = WMMA_F16(a[mi], b[ni], acc[mi][ni]);
  }

  if constexpr (HEADT) {
    // out[(b*8+h)*64 + d][n] ; within a lane g walks consecutive n.
#pragma unroll
    for (int ni = 0; ni < 4; ++ni) {
      const int col = n0 + ni * 16 + l15; // h*64 + d
      const int hh = col >> 6, dd = col & 63;
      const float bv = bias[col];
#pragma unroll
      for (int mi = 0; mi < 4; ++mi) {
        const int nbase = mi * 16 + half * 8; // n within batch row-block
        v8h hv;
#pragma unroll
        for (int g = 0; g < 8; ++g) hv[g] = (_Float16)(acc[mi][ni][g] + bv);
        const size_t off = ((size_t)((m0 >> 6) * Hh + hh) * 64 + dd) * 64 + nbase;
        *(v8h*)((_Float16*)C + off) = hv;
      }
    }
  } else {
#pragma unroll
    for (int ni = 0; ni < 4; ++ni) {
      const int col = n0 + ni * 16 + l15;
      const float bv = bias[col];
#pragma unroll
      for (int mi = 0; mi < 4; ++mi) {
#pragma unroll
        for (int g = 0; g < 8; ++g) {
          const int row = m0 + mi * 16 + g + half * 8;
          C[(size_t)row * N + col] = (OT)(acc[mi][ni][g] + bv);
        }
      }
    }
  }
}

// ---------------------------------------------------------------------------
// Attention: one wave per (b,h), 4 waves per block.
// S = Q^T K * scale  (fragments straight from qT/kT, all vector loads)
// softmax over e; P -> LDS (scalar transpose-store, the only scalar path)
// ctx^T = P V^T      (A from LDS b128, B from global b128, v8h stores)
// ---------------------------------------------------------------------------
__global__ __launch_bounds__(128) void attn_kernel(const _Float16* __restrict__ qT,
                                                   const _Float16* __restrict__ kT,
                                                   const _Float16* __restrict__ v_h,
                                                   _Float16* __restrict__ ctx_h) {
  __shared__ alignas(16) _Float16 sP[4][64][72];

  const int lane = threadIdx.x & 31;
  const int wid  = threadIdx.x >> 5;
  const int l15  = lane & 15;
  const int half = lane >> 4;
  const int bh = blockIdx.x * 4 + wid;
  const int b  = bh >> 3;
  const int h  = bh & 7;

  const _Float16* qTb = qT + (size_t)bh * 64 * 64; // [d][n]
  const _Float16* kTb = kT + (size_t)bh * 64 * 64; // [e][n]
  const _Float16* vb  = v_h + (size_t)b * Nn * DIM + h * HD; // [n][e] stride DIM

  const v8f zero8 = {0.f, 0.f, 0.f, 0.f, 0.f, 0.f, 0.f, 0.f};

  // ---- S[d][e] = sum_n Q[n][d] K[n][e] -------------------------------------
  v8f s[4][4];
#pragma unroll
  for (int i = 0; i < 4; ++i)
#pragma unroll
    for (int j = 0; j < 4; ++j) s[i][j] = zero8;

#pragma unroll
  for (int ks = 0; ks < 2; ++ks) {
    const int kbA = ks * 32 + half * 16; // A elems 0-15 -> n=kbA..
    const int kbB = ks * 32 + half * 16; // B elems 0-15 -> n=kbB..
    v16h aq[4], bk[4];
#pragma unroll
    for (int di = 0; di < 4; ++di) {
      const _Float16* p = qTb + (size_t)(di * 16 + l15) * 64 + kbA;
      v8h h0 = *(const v8h*)(p);
      v8h h1 = *(const v8h*)(p + 8);
#pragma unroll
      for (int j = 0; j < 8; ++j) { aq[di][j] = h0[j]; aq[di][8 + j] = h1[j]; }
    }
#pragma unroll
    for (int ei = 0; ei < 4; ++ei) {
      const _Float16* p = kTb + (size_t)(ei * 16 + l15) * 64 + kbB;
      v8h w0 = *(const v8h*)(p);
      v8h w1 = *(const v8h*)(p + 8);
#pragma unroll
      for (int j = 0; j < 8; ++j) { bk[ei][j] = w0[j]; bk[ei][8 + j] = w1[j]; }
    }
#pragma unroll
    for (int di = 0; di < 4; ++di)
#pragma unroll
      for (int ei = 0; ei < 4; ++ei)
        s[di][ei] = WMMA_F16(aq[di], bk[ei], s[di][ei]);
  }

  // ---- softmax over e; row d = 16*di + g + 8*half lives in one 16-lane half.
  const float scale = 0.125f; // 1/sqrt(N=64)
#pragma unroll
  for (int di = 0; di < 4; ++di) {
#pragma unroll
    for (int g = 0; g < 8; ++g) {
      float x[4];
#pragma unroll
      for (int ei = 0; ei < 4; ++ei) x[ei] = s[di][ei][g] * scale;
      float mx = fmaxf(fmaxf(x[0], x[1]), fmaxf(x[2], x[3]));
#pragma unroll
      for (int off = 1; off < 16; off <<= 1) mx = fmaxf(mx, __shfl_xor(mx, off));
      float sum = 0.f;
#pragma unroll
      for (int ei = 0; ei < 4; ++ei) { x[ei] = __expf(x[ei] - mx); sum += x[ei]; }
#pragma unroll
      for (int off = 1; off < 16; off <<= 1) sum += __shfl_xor(sum, off);
      const float rs = 1.f / sum;
      const int d = di * 16 + g + half * 8;
#pragma unroll
      for (int ei = 0; ei < 4; ++ei)
        sP[wid][d][ei * 16 + l15] = (_Float16)(x[ei] * rs);
    }
  }
  __syncthreads();

  // ---- ctx^T[d][n] = sum_e P[d][e] V[n][e]  (A = P, B = V^T) ---------------
  v8f c[4][4];
#pragma unroll
  for (int i = 0; i < 4; ++i)
#pragma unroll
    for (int j = 0; j < 4; ++j) c[i][j] = zero8;

#pragma unroll
  for (int ks = 0; ks < 2; ++ks) {
    const int kbA = ks * 32 + half * 16;
    const int kbB = ks * 32 + half * 16;
    v16h ap[4], bv[4];
#pragma unroll
    for (int mi = 0; mi < 4; ++mi) { // rows d
      const int d = mi * 16 + l15;
      v8h h0 = *(const v8h*)&sP[wid][d][kbA];
      v8h h1 = *(const v8h*)&sP[wid][d][kbA + 8];
#pragma unroll
      for (int j = 0; j < 8; ++j) { ap[mi][j] = h0[j]; ap[mi][8 + j] = h1[j]; }
    }
#pragma unroll
    for (int ni = 0; ni < 4; ++ni) { // cols n
      const _Float16* p = vb + (size_t)(ni * 16 + l15) * DIM + kbB;
      v8h p0 = *(const v8h*)(p);
      v8h p1 = *(const v8h*)(p + 8);
#pragma unroll
      for (int j = 0; j < 8; ++j) { bv[ni][j] = p0[j]; bv[ni][8 + j] = p1[j]; }
    }
#pragma unroll
    for (int mi = 0; mi < 4; ++mi)
#pragma unroll
      for (int ni = 0; ni < 4; ++ni)
        c[mi][ni] = WMMA_F16(ap[mi], bv[ni], c[mi][ni]);
  }

  // ctx[(b*64+n)*512 + h*64 + d]; g walks consecutive d -> v8h stores.
  _Float16* cb = ctx_h + (size_t)b * Nn * DIM + h * HD;
#pragma unroll
  for (int ni = 0; ni < 4; ++ni) {
    const int n = ni * 16 + l15;
#pragma unroll
    for (int mi = 0; mi < 4; ++mi) {
      const int dbase = mi * 16 + half * 8;
      v8h hv;
#pragma unroll
      for (int g = 0; g < 8; ++g) hv[g] = (_Float16)c[mi][ni][g];
      *(v8h*)&cb[(size_t)n * DIM + dbase] = hv;
    }
  }
}

// ---------------------------------------------------------------------------
extern "C" void kernel_launch(void* const* d_in, const int* in_sizes, int n_in,
                              void* d_out, int out_size, void* d_ws, size_t ws_size,
                              hipStream_t stream) {
  const float* x      = (const float*)d_in[0];
  const float* enc    = (const float*)d_in[1];
  const float* q_w    = (const float*)d_in[2];
  const float* q_b    = (const float*)d_in[3];
  const float* kv_w   = (const float*)d_in[4];
  const float* kv_b   = (const float*)d_in[5];
  const float* proj_w = (const float*)d_in[6];
  const float* proj_b = (const float*)d_in[7];
  float* out = (float*)d_out;

  char* ws = (char*)d_ws;
  const size_t wsz = (size_t)DIM * DIM * sizeof(_Float16);   // 512 KB per weight
  const size_t big = (size_t)Mrows * DIM * sizeof(_Float16); // 32 MB per tensor
  _Float16* qwT = (_Float16*)(ws + 0 * wsz);
  _Float16* kwT = (_Float16*)(ws + 1 * wsz);
  _Float16* vwT = (_Float16*)(ws + 2 * wsz);
  _Float16* pwT = (_Float16*)(ws + 3 * wsz);
  _Float16* qTt = (_Float16*)(ws + 4 * wsz);            // [b,h][d][n]
  _Float16* kTt = (_Float16*)(ws + 4 * wsz + 1 * big);  // [b,h][e][n]
  _Float16* v_h = (_Float16*)(ws + 4 * wsz + 2 * big);  // [b,n][h*64+e]
  _Float16* ctx = (_Float16*)(ws + 4 * wsz + 3 * big);  // [b,n][h*64+d]

  prep_weights<<<(DIM * DIM) / 256, 256, 0, stream>>>(q_w, kv_w, proj_w,
                                                      qwT, kwT, vwT, pwT);

  const int blocks = Mrows / 64; // one 64-row A panel per block, 8 n-tiles

  gemm64<float, _Float16, true ><<<blocks, 256, 0, stream>>>(x,   qwT, q_b,        qTt, Mrows, DIM, DIM);
  gemm64<float, _Float16, true ><<<blocks, 256, 0, stream>>>(enc, kwT, kv_b,       kTt, Mrows, DIM, DIM);
  gemm64<float, _Float16, false><<<blocks, 256, 0, stream>>>(enc, vwT, kv_b + DIM, v_h, Mrows, DIM, DIM);

  attn_kernel<<<(Bb * Hh) / 4, 128, 0, stream>>>(qTt, kTt, v_h, ctx);

  gemm64<_Float16, float, false><<<blocks, 256, 0, stream>>>(ctx, pwT, proj_b, out, Mrows, DIM, DIM);
}
